// PGJANET_23545010717403
// MI455X (gfx1250) — matrix operand order repaired and checked by reading, not compiled, not run on hardware
//
#include <hip/hip_runtime.h>

// PGJANET recurrent network, B=128, T=1024, H=256, O=2.
// Grid of 8 workgroups; each owns a 16-row batch slice and runs the full
// T=1024 recurrence. Gate GEMMs use v_wmma_f32_16x16x32_f16; h/u state lives
// in LDS (f32 master + f16 mirror); weights are repacked once to an f16 WMMA
// B-fragment layout in d_ws and streamed from L2 every timestep (an opaque
// zero offset prevents LLVM from hoisting+spilling the loads while keeping
// them in the global address space -> global_load_b128, LOADcnt only).
// Per-timestep inputs are double-buffered into LDS with
// global_load_async_to_lds_b64 / s_wait_asynccnt.

#define BB 128
#define TT 1024
#define HH 256
#define OO 2
#define KTILES 8        // K = 256 / 32
#define NTILES 16       // N = 256 / 16
#define H16P 272        // f16 row stride (halves): 544B, 16B-aligned, breaks bank conflicts
#define H32P 260        // f32 row stride (floats)

typedef __attribute__((ext_vector_type(16))) _Float16 v16h;
typedef __attribute__((ext_vector_type(8)))  _Float16 v8h;
typedef __attribute__((ext_vector_type(8)))  float    v8f;

union HFrag { v16h v; v8h h[2]; };

__device__ __forceinline__ float fexp2(float x) { return __builtin_amdgcn_exp2f(x); }
__device__ __forceinline__ float frcp(float x)  { return __builtin_amdgcn_rcpf(x); }
__device__ __forceinline__ float ftanh(float x) {
  // tanh(x) = 1 - 2/(exp(2x)+1), exp(2x) = exp2(2*log2(e)*x)
  float s = fexp2(x * 2.885390082f);
  return 1.0f - 2.0f * frcp(s + 1.0f);
}
__device__ __forceinline__ float fsig(float x) {
  return frcp(1.0f + fexp2(x * -1.442695041f));
}

// ---------------------------------------------------------------------------
// Weight repack: f32 row-major W[j][k] slices -> f16 WMMA B-fragment tiles.
// Tile (g, kt, nt): 32 lanes x 16 halves, lane-contiguous.
//   lane <16 : column n0+lane,    K = k0+{0..7, 16..23}
//   lane>=16 : column n0+lane-16, K = k0+{8..15, 24..31}
// ---------------------------------------------------------------------------
__global__ void pgjanet_pack(const float* __restrict__ Wa, const float* __restrict__ Wc,
                             const float* __restrict__ Ws, const float* __restrict__ Wf,
                             const float* __restrict__ Wu, _Float16* __restrict__ pw)
{
  int t = blockIdx.x * blockDim.x + threadIdx.x;   // 7 * 8 * 16 * 32 = 28672 tasks
  if (t >= 7 * KTILES * NTILES * 32) return;
  int lane = t & 31;
  int nt   = (t >> 5) & 15;
  int kt   = (t >> 9) & 7;
  int g    = t >> 12;                               // 0..6: a, c, s, f_h, u_h, f_u, u_u
  const float* src; int ld, koff;
  switch (g) {
    case 0:  src = Wa; ld = HH + 1; koff = 0;  break;
    case 1:  src = Wc; ld = HH + 1; koff = 0;  break;
    case 2:  src = Ws; ld = HH + 1; koff = 0;  break;
    case 3:  src = Wf; ld = 2 * HH; koff = 0;  break;
    case 4:  src = Wu; ld = 2 * HH; koff = 0;  break;
    case 5:  src = Wf; ld = 2 * HH; koff = HH; break;
    default: src = Wu; ld = 2 * HH; koff = HH; break;
  }
  int n  = nt * 16 + (lane & 15);
  int kb = kt * 32 + ((lane & 16) ? 8 : 0) + koff;
  _Float16* dst = pw + (size_t)(((g * KTILES + kt) * NTILES + nt) * 512 + lane * 16);
  const float* s = src + (size_t)n * ld + kb;
#pragma unroll
  for (int e = 0; e < 16; ++e) dst[e] = (_Float16)s[e + (e & 8)];
}

// ---------------------------------------------------------------------------
// Main recurrence kernel: 8 blocks x 512 threads (16 waves). Wave w owns
// N-tile w (columns 16w..16w+15) of every gate.
// ---------------------------------------------------------------------------
__global__ void __launch_bounds__(512, 1)
pgjanet_rnn(const float* __restrict__ x, const _Float16* __restrict__ pw,
            const float* __restrict__ Wa, const float* __restrict__ Wc,
            const float* __restrict__ Ws,
            const float* __restrict__ ba, const float* __restrict__ bc,
            const float* __restrict__ bs, const float* __restrict__ bf,
            const float* __restrict__ bu,
            const float* __restrict__ Wo, const float* __restrict__ bo,
            float* __restrict__ out)
{
  __shared__ float    sWo[OO * HH];
  __shared__ float    sWx[3 * HH];      // wa_x, wc_x, ws_x columns
  __shared__ float    sB[5 * HH];       // ba, bc, bs, bf, bu
  __shared__ float    sFeat[16 * 4];    // amp, cos, sin per batch row
  __shared__ float    xstage[2][32];    // async-staged x[:, t, :] double buffer
  __shared__ float    h32[16 * H32P];   // f32 master h
  __shared__ _Float16 h16[16 * H16P];   // f16 mirror for WMMA A fragments
  __shared__ _Float16 u16s[16 * H16P];  // f16 u for phase-2 A fragments

  const int tid  = threadIdx.x;
  const int lane = tid & 31;
  const int w    = tid >> 5;            // wave id == n-tile id, 0..15
  const int r0   = blockIdx.x * 16;     // batch-row base

  // ---- one-time LDS staging ----
  for (int i = tid; i < OO * HH; i += 512) sWo[i] = Wo[i];
  for (int i = tid; i < HH; i += 512) {
    sWx[i]          = Wa[(size_t)i * (HH + 1) + HH];
    sWx[HH + i]     = Wc[(size_t)i * (HH + 1) + HH];
    sWx[2 * HH + i] = Ws[(size_t)i * (HH + 1) + HH];
    sB[i]           = ba[i];
    sB[HH + i]      = bc[i];
    sB[2 * HH + i]  = bs[i];
    sB[3 * HH + i]  = bf[i];
    sB[4 * HH + i]  = bu[i];
  }
  for (int i = tid; i < 16 * H32P; i += 512) h32[i] = 0.0f;
  for (int i = tid; i < 16 * H16P; i += 512) {
    h16[i]  = (_Float16)0.0f;
    u16s[i] = (_Float16)0.0f;
  }

  // ---- stage x[:, 0, :] asynchronously into LDS (wave 0, lanes dup-16) ----
  if (w == 0) {
    int row = lane & 15;
    const float* gp = &x[((size_t)(r0 + row) * TT + 0) * 2];
    unsigned lds = (unsigned)(size_t)&xstage[0][row * 2];
    asm volatile("global_load_async_to_lds_b64 %0, %1, off"
                 :: "v"(lds), "v"(gp) : "memory");
  }
  __syncthreads();

  // A-fragment addressing (16-bit A 16x32 swizzle):
  //   lanes 0-15:  K runs {0..7, 16..23};  lanes 16-31: K runs {8..15, 24..31}
  const int arow  = lane & 15;
  const int akoff = (lane & 16) ? 8 : 0;
  // C/D layout: VGPR r, lane<16 -> M=r, N=lane; lane>=16 -> M=8+r, N=lane-16
  const int ncol  = w * 16 + (lane & 15);
  const int mbase = (lane & 16) ? 8 : 0;

  const v8f vzero = {0.f, 0.f, 0.f, 0.f, 0.f, 0.f, 0.f, 0.f};

  for (int t = 0; t < TT; ++t) {
    // Opaque ZERO offset: blocks loop-invariant hoisting (and scratch spills)
    // of the per-timestep B-fragment loads, while keeping the base pointer's
    // global address space (emits global_load_b128, not flat_load_b128).
    size_t tzero = 0;
    asm volatile("" : "+s"(tzero));
    const _Float16* pwt = pw + tzero;

    // ---- consume staged x, compute features, prefetch-stage t+1 ----
    if (w == 0) {
      asm volatile("s_wait_asynccnt 0x0" ::: "memory");  // xstage[t&1] landed
    }
    if (tid < 16) {
      float iv  = xstage[t & 1][tid * 2 + 0];
      float qv  = xstage[t & 1][tid * 2 + 1];
      float amp = __builtin_amdgcn_sqrtf(iv * iv + qv * qv);
      float ra  = (amp > 0.0f) ? frcp(amp) : 0.0f;
      sFeat[tid * 4 + 0] = amp;
      sFeat[tid * 4 + 1] = (amp > 0.0f) ? iv * ra : 1.0f;  // cos(atan2(q,i))
      sFeat[tid * 4 + 2] = (amp > 0.0f) ? qv * ra : 0.0f;  // sin(atan2(q,i))
    }
    if (w == 0) {
      int tn  = (t + 1 < TT) ? (t + 1) : (TT - 1);
      int row = lane & 15;
      const float* gp = &x[((size_t)(r0 + row) * TT + tn) * 2];
      unsigned lds = (unsigned)(size_t)&xstage[(t + 1) & 1][row * 2];
      asm volatile("global_load_async_to_lds_b64 %0, %1, off"
                   :: "v"(lds), "v"(gp) : "memory");
    }
    __syncthreads();  // b1: feats + previous h16 visible

    // ---- phase 1: pre[g] = h @ Wg^T for g in {a,c,s,f_h,u_h} ----
    v8f acc[5];
#pragma unroll
    for (int g = 0; g < 5; ++g) acc[g] = vzero;

#pragma unroll
    for (int kt = 0; kt < KTILES; ++kt) {
      HFrag af;
      const _Float16* hp = &h16[arow * H16P + kt * 32 + akoff];
      af.h[0] = *(const v8h*)(hp);
      af.h[1] = *(const v8h*)(hp + 16);
#pragma unroll
      for (int g = 0; g < 5; ++g) {
        const _Float16* bp =
            pwt + (size_t)(((g * KTILES + kt) * NTILES + w) * 512 + lane * 16);
        HFrag bfm;
        bfm.h[0] = *(const v8h*)(bp);
        bfm.h[1] = *(const v8h*)(bp + 8);
        acc[g] = __builtin_amdgcn_wmma_f32_16x16x32_f16(
            false, af.v, false, bfm.v, (short)0, acc[g], false, false);
      }
    }

    // ---- elementwise: a, c, s gates and u ----
    const float wax = sWx[ncol], wcx = sWx[HH + ncol], wsx = sWx[2 * HH + ncol];
    const float bav = sB[ncol], bcv = sB[HH + ncol], bsv = sB[2 * HH + ncol];
    v8f accF = acc[3];
    v8f accV = acc[4];
#pragma unroll
    for (int r = 0; r < 8; ++r) {
      int m = mbase + r;
      float a = ftanh(acc[0][r] + sFeat[m * 4 + 0] * wax + bav);
      float c = ftanh(acc[1][r] + sFeat[m * 4 + 1] * wcx + bcv);
      float s = ftanh(acc[2][r] + sFeat[m * 4 + 2] * wsx + bsv);
      float uu = a * c * s * (1.0f - a) * (1.0f - c) * (1.0f - s);
      u16s[m * H16P + ncol] = (_Float16)uu;
    }
    __syncthreads();  // b2: u16 ready, all phase-1 reads of h16 done

    // ---- phase 2: accF += u @ Wf_u^T ; accV += u @ Wu_u^T ----
#pragma unroll
    for (int kt = 0; kt < KTILES; ++kt) {
      HFrag af;
      const _Float16* up = &u16s[arow * H16P + kt * 32 + akoff];
      af.h[0] = *(const v8h*)(up);
      af.h[1] = *(const v8h*)(up + 16);
      const _Float16* bpf =
          pwt + (size_t)(((5 * KTILES + kt) * NTILES + w) * 512 + lane * 16);
      const _Float16* bpv =
          pwt + (size_t)(((6 * KTILES + kt) * NTILES + w) * 512 + lane * 16);
      HFrag bff, bvv;
      bff.h[0] = *(const v8h*)(bpf); bff.h[1] = *(const v8h*)(bpf + 8);
      bvv.h[0] = *(const v8h*)(bpv); bvv.h[1] = *(const v8h*)(bpv + 8);
      accF = __builtin_amdgcn_wmma_f32_16x16x32_f16(
          false, af.v, false, bff.v, (short)0, accF, false, false);
      accV = __builtin_amdgcn_wmma_f32_16x16x32_f16(
          false, af.v, false, bvv.v, (short)0, accV, false, false);
    }

    // ---- h update ----
    const float bfv = sB[3 * HH + ncol], buv = sB[4 * HH + ncol];
#pragma unroll
    for (int r = 0; r < 8; ++r) {
      int m = mbase + r;
      float f    = fsig(accF[r] + bfv);
      float v    = ftanh(accV[r] + buv);
      float hold = h32[m * H32P + ncol];
      float hn   = f * hold + (1.0f - f) * v;
      h32[m * H32P + ncol] = hn;
      h16[m * H16P + ncol] = (_Float16)hn;
    }
    __syncthreads();  // b3: new h visible

    // ---- fused output projection: out[b,t,o] = h . Wo[o] + bo[o] ----
    if (tid < 32) {
      int b = tid & 15, o = tid >> 4;
      float accO = 0.0f;
#pragma unroll 4
      for (int c = 0; c < HH; ++c) accO += h32[b * H32P + c] * sWo[o * HH + c];
      out[((size_t)(r0 + b) * TT + t) * OO + o] = accO + bo[o];
    }
    // (next iteration's b1 separates these h32 reads from the next overwrite)
  }
}

// ---------------------------------------------------------------------------
// Harness entry. Input order: x, Wa, ba, Wc, bc, Ws, bs, Wf, bf, Wu, bu, Wo, bo
// ---------------------------------------------------------------------------
extern "C" void kernel_launch(void* const* d_in, const int* in_sizes, int n_in,
                              void* d_out, int out_size, void* d_ws, size_t ws_size,
                              hipStream_t stream) {
  const float* x  = (const float*)d_in[0];
  const float* Wa = (const float*)d_in[1];
  const float* ba = (const float*)d_in[2];
  const float* Wc = (const float*)d_in[3];
  const float* bc = (const float*)d_in[4];
  const float* Ws = (const float*)d_in[5];
  const float* bs = (const float*)d_in[6];
  const float* Wf = (const float*)d_in[7];
  const float* bf = (const float*)d_in[8];
  const float* Wu = (const float*)d_in[9];
  const float* bu = (const float*)d_in[10];
  const float* Wo = (const float*)d_in[11];
  const float* bo = (const float*)d_in[12];
  float* out = (float*)d_out;

  _Float16* pw = (_Float16*)d_ws;  // needs 7*256*256*2 = 917504 bytes

  // Repack weights to f16 WMMA fragment layout (re-run every call: deterministic).
  int packTasks = 7 * KTILES * NTILES * 32;
  pgjanet_pack<<<(packTasks + 255) / 256, 256, 0, stream>>>(Wa, Wc, Ws, Wf, Wu, pw);

  // One workgroup per 16-row batch slice.
  pgjanet_rnn<<<BB / 16, 512, 0, stream>>>(x, pw, Wa, Wc, Ws,
                                           ba, bc, bs, bf, bu, Wo, bo, out);
}
